// GNNModel_51986284151377
// MI455X (gfx1250) — matrix operand order repaired and checked by reading, not compiled, not run on hardware
//
#include <hip/hip_runtime.h>
#include <hip/hip_bf16.h>
#include <cstdint>

#define N_NODES 50000
#define N_EDGES 300000
#define HDIM    128
#define NLAYER  6

typedef __attribute__((ext_vector_type(16))) __bf16       v16bf;
typedef __attribute__((ext_vector_type(8)))  float        v8f;
typedef __attribute__((ext_vector_type(8)))  unsigned int v8u;

__device__ __forceinline__ unsigned short f2bf(float f) {
  unsigned int u = __builtin_bit_cast(unsigned int, f);
  u += 0x7fffu + ((u >> 16) & 1u);            // round-to-nearest-even
  return (unsigned short)(u >> 16);
}

__device__ __forceinline__ v16bf frag_from(uint4 lo, uint4 hi) {
  v8u t;
  t[0]=lo.x; t[1]=lo.y; t[2]=lo.z; t[3]=lo.w;
  t[4]=hi.x; t[5]=hi.y; t[6]=hi.z; t[7]=hi.w;
  return __builtin_bit_cast(v16bf, t);
}

// A fragment (16x32 bf16) from an LDS row-major tile, per ISA layout:
// lanes 0-15: row M=lane, K = kt*32 + {0..7, 16..23}; lanes 16-31: K += 8.
__device__ __forceinline__ v16bf a_frag_lds(const unsigned short* sA, int ld, int kt, int lane) {
  int row = lane & 15;
  int ko  = (lane >> 4) << 3;                 // 0 or 8
  const unsigned short* p = sA + row * ld + kt * 32 + ko;
  uint4 lo = *(const uint4*)p;
  uint4 hi = *(const uint4*)(p + 16);
  return frag_from(lo, hi);
}

// B fragment from pre-swizzled global weights: lane holds 16 contiguous bf16.
__device__ __forceinline__ v16bf b_frag_glb(const unsigned short* wfrag, int lane) {
  const uint4* p = (const uint4*)(wfrag + lane * 16);
  return frag_from(p[0], p[1]);
}

#define WMMA_BF16(acc, a, b)                                                     \
  acc = __builtin_amdgcn_wmma_f32_16x16x32_bf16(false, (a), false, (b),          \
                                                (short)0, (acc), false, false)

// Async Global -> LDS copy (CDNA5 GLOBAL_LOAD_ASYNC_TO_LDS_B128, ASYNCcnt).
// LDS destination is the low 32 bits of the generic shared pointer (LDS
// aperture keeps the workgroup LDS byte offset in addr[31:0]).
__device__ __forceinline__ void async_ld_b128(const unsigned short* lds_dst,
                                              const unsigned short* gsrc) {
  unsigned off = (unsigned)(uintptr_t)lds_dst;
  asm volatile("global_load_async_to_lds_b128 %0, %1, off"
               :: "v"(off), "v"(gsrc)
               : "memory");
}
__device__ __forceinline__ void wait_async_all() {
  asm volatile("s_wait_asynccnt 0x0" ::: "memory");
}

// ---------------------------------------------------------------- utilities
__global__ void zero_f32(float* __restrict__ p, size_t n) {
  size_t i = (size_t)blockIdx.x * blockDim.x + threadIdx.x;
  if (i < n) p[i] = 0.f;
}

// Convert fp32 KxN weight -> bf16, padded to Kp x Np, swizzled into WMMA
// B-fragment order: frag(kt,nt) is 512 ushorts; element (lane,i) maps to
// K = kt*32 + (i&7) + (i>=8)*16 + (lane>=16)*8, N = nt*16 + (lane&15).
__global__ void conv_w(const float* __restrict__ src, unsigned short* __restrict__ dst,
                       int K, int Nn, int Kp, int Np) {
  int t = blockIdx.x * blockDim.x + threadIdx.x;
  int total = Kp * Np;
  if (t >= total) return;
  int ntiles = Np >> 4;
  int frag = t >> 9;
  int pos  = t & 511;
  int lane = pos >> 4;
  int i    = pos & 15;
  int kt = frag / ntiles, nt = frag % ntiles;
  int k = kt * 32 + (i & 7) + ((i >> 3) << 4) + ((lane >> 4) << 3);
  int n = nt * 16 + (lane & 15);
  float val = (k < K && n < Nn) ? src[(size_t)k * Nn + n] : 0.f;
  dst[t] = f2bf(val);
}

__global__ void deg_kernel(const int* __restrict__ ei, float* __restrict__ deg) {
  int e = blockIdx.x * blockDim.x + threadIdx.x;
  if (e < N_EDGES) atomicAdd(&deg[ei[N_EDGES + e]], 1.0f);
}

// ---------------------------------------------------------------- embedding
// feat = [r | v(6) | gf(12) | pad] (19 -> 32), h = relu(relu(feat@W1+b1)@W2+b2)
__global__ __launch_bounds__(256) void embed_kernel(
    const float* __restrict__ r, const float* __restrict__ v,
    const float* __restrict__ dom, const float* __restrict__ t,
    const float* __restrict__ xg, const float* __restrict__ domn, const float* __restrict__ tn,
    const unsigned short* __restrict__ W1, const float* __restrict__ b1,
    const unsigned short* __restrict__ W2, const float* __restrict__ b2,
    float* __restrict__ h, unsigned short* __restrict__ hbf)
{
  __shared__ __align__(16) unsigned short sA[16 * 32];
  __shared__ __align__(16) unsigned short sMid[16 * 128];
  const int tid = threadIdx.x;
  const int row = tid >> 4, q = tid & 15;
  const int n0 = blockIdx.x * 16;
  const int node = n0 + row;
  if (q == 0) {
    float ext[32];
#pragma unroll
    for (int i = 0; i < 32; ++i) ext[i] = 0.f;
    if (node < N_NODES) {
      ext[0] = r[node];
#pragma unroll
      for (int j = 0; j < 6; ++j) ext[1 + j] = v[node * 6 + j];
      ext[7]=dom[0]; ext[8]=dom[1]; ext[9]=dom[2]; ext[10]=t[0];
      ext[11]=xg[0]; ext[12]=xg[1]; ext[13]=xg[2]; ext[14]=xg[3];
      ext[15]=domn[0]; ext[16]=domn[1]; ext[17]=domn[2]; ext[18]=tn[0];
    }
#pragma unroll
    for (int i = 0; i < 32; ++i) sA[row * 32 + i] = f2bf(ext[i]);
  }
  __syncthreads();
  const int wave = tid >> 5, lane = tid & 31;
  const int col = wave * 16 + (lane & 15);
  v8f acc = {};
  {
    v16bf a = a_frag_lds(sA, 32, 0, lane);
    v16bf b = b_frag_glb(W1 + (size_t)wave * 512, lane);
    WMMA_BF16(acc, a, b);
  }
  float bias = b1[col];
#pragma unroll
  for (int i = 0; i < 8; ++i) {
    int rr = i + ((lane >> 4) << 3);
    sMid[rr * 128 + col] = f2bf(fmaxf(acc[i] + bias, 0.f));
  }
  __syncthreads();
  v8f acc2 = {};
  for (int kt = 0; kt < 4; ++kt) {
    v16bf a = a_frag_lds(sMid, 128, kt, lane);
    v16bf b = b_frag_glb(W2 + (size_t)(kt * 8 + wave) * 512, lane);
    WMMA_BF16(acc2, a, b);
  }
  float bias2 = b2[col];
#pragma unroll
  for (int i = 0; i < 8; ++i) {
    int rr = i + ((lane >> 4) << 3);
    int nd = n0 + rr;
    if (nd < N_NODES) {
      float val = fmaxf(acc2[i] + bias2, 0.f);
      h[(size_t)nd * HDIM + col]   = val;
      hbf[(size_t)nd * HDIM + col] = f2bf(val);
    }
  }
}

// ---------------------------------------------------------------- edge MLP
// Fused: async-gather m_in(16x288 bf16) into LDS -> GEMM1(relu) ->
// GEMM2(relu) -> /deg -> atomic scatter into agg[dst].
__global__ __launch_bounds__(256) void edge_kernel(
    const int* __restrict__ ei,
    const unsigned short* __restrict__ hbf,
    const float* __restrict__ pos, const float* __restrict__ v, const float* __restrict__ r,
    const float* __restrict__ dom, const float* __restrict__ t, const float* __restrict__ xg,
    const float* __restrict__ domn, const float* __restrict__ tn,
    const unsigned short* __restrict__ W1, const float* __restrict__ b1,
    const unsigned short* __restrict__ W2, const float* __restrict__ b2,
    const float* __restrict__ deg, float* __restrict__ agg)
{
  __shared__ __align__(16) unsigned short sA[16 * 288];
  __shared__ __align__(16) unsigned short sMid[16 * 128];
  __shared__ int   sSrc[16];
  __shared__ int   sDst[16];
  __shared__ float sInv[16];
  const int tid = threadIdx.x;
  const int e0 = blockIdx.x * 16;
  if (tid < 16) {
    int e = e0 + tid;
    int s = 0, d = 0;
    if (e < N_EDGES) { s = ei[e]; d = ei[N_EDGES + e]; }
    sSrc[tid] = s; sDst[tid] = d;
    sInv[tid] = 1.0f / fmaxf(deg[d], 1.0f);
  }
  __syncthreads();
  const int row = tid >> 4, q = tid & 15;
  const int e = e0 + row;
  const int sI = sSrc[row], dI = sDst[row];
  // h[dst] -> cols [0,128), h[src] -> cols [128,256): direct L2 -> LDS.
  async_ld_b128(&sA[row * 288 + q * 8],       hbf + (size_t)dI * HDIM + q * 8);
  async_ld_b128(&sA[row * 288 + 128 + q * 8], hbf + (size_t)sI * HDIM + q * 8);
  if (q == 0) {   // cols 256..287: dpos(3) dv(6) r_dst r_src gf(12) pad
    float ext[32];
#pragma unroll
    for (int i = 0; i < 32; ++i) ext[i] = 0.f;
    if (e < N_EDGES) {
#pragma unroll
      for (int j = 0; j < 3; ++j) {
        float D  = dom[j];
        float dp = pos[dI * 3 + j] - pos[sI * 3 + j];
        dp -= D * rintf(dp / D);
        ext[j] = dp;
      }
#pragma unroll
      for (int j = 0; j < 6; ++j) ext[3 + j] = v[dI * 6 + j] - v[sI * 6 + j];
      ext[9]  = r[dI];
      ext[10] = r[sI];
      ext[11]=dom[0]; ext[12]=dom[1]; ext[13]=dom[2]; ext[14]=t[0];
      ext[15]=xg[0]; ext[16]=xg[1]; ext[17]=xg[2]; ext[18]=xg[3];
      ext[19]=domn[0]; ext[20]=domn[1]; ext[21]=domn[2]; ext[22]=tn[0];
    }
#pragma unroll
    for (int i = 0; i < 32; ++i) sA[row * 288 + 256 + i] = f2bf(ext[i]);
  }
  wait_async_all();
  __syncthreads();
  const int wave = tid >> 5, lane = tid & 31;
  const int col = wave * 16 + (lane & 15);
  v8f acc = {};
  for (int kt = 0; kt < 9; ++kt) {
    __builtin_prefetch(W1 + (size_t)((kt + 1) * 8 + wave) * 512, 0, 1);
    v16bf a = a_frag_lds(sA, 288, kt, lane);
    v16bf b = b_frag_glb(W1 + (size_t)(kt * 8 + wave) * 512, lane);
    WMMA_BF16(acc, a, b);
  }
  float bias = b1[col];
#pragma unroll
  for (int i = 0; i < 8; ++i) {
    int rr = i + ((lane >> 4) << 3);
    sMid[rr * 128 + col] = f2bf(fmaxf(acc[i] + bias, 0.f));
  }
  __syncthreads();
  v8f acc2 = {};
  for (int kt = 0; kt < 4; ++kt) {
    v16bf a = a_frag_lds(sMid, 128, kt, lane);
    v16bf b = b_frag_glb(W2 + (size_t)(kt * 8 + wave) * 512, lane);
    WMMA_BF16(acc2, a, b);
  }
  float bias2 = b2[col];
#pragma unroll
  for (int i = 0; i < 8; ++i) {
    int rr = i + ((lane >> 4) << 3);
    if (e0 + rr < N_EDGES) {
      float m = fmaxf(acc2[i] + bias2, 0.f) * sInv[rr];
      atomicAdd(&agg[(size_t)sDst[rr] * HDIM + col], m);
    }
  }
}

// ---------------------------------------------------------------- node MLP
// u_in = [h | agg | gf | pad] (288), upd = relu(u@W1+b1)@W2+b2, hn = h+upd,
// plus per-channel sum/sumsq for instance norm.
__global__ __launch_bounds__(256) void node_kernel(
    const unsigned short* __restrict__ hbf, const float* __restrict__ h,
    const float* __restrict__ agg,
    const float* __restrict__ dom, const float* __restrict__ t, const float* __restrict__ xg,
    const float* __restrict__ domn, const float* __restrict__ tn,
    const unsigned short* __restrict__ W1, const float* __restrict__ b1,
    const unsigned short* __restrict__ W2, const float* __restrict__ b2,
    float* __restrict__ htmp, float* __restrict__ sum, float* __restrict__ sumsq)
{
  __shared__ __align__(16) unsigned short sA[16 * 288];
  __shared__ __align__(16) unsigned short sMid[16 * 128];
  const int tid = threadIdx.x;
  const int row = tid >> 4, q = tid & 15;
  const int n0 = blockIdx.x * 16;
  const int node = n0 + row;
  const bool ok = node < N_NODES;
  const int nodeC = ok ? node : 0;
  async_ld_b128(&sA[row * 288 + q * 8], hbf + (size_t)nodeC * HDIM + q * 8);
#pragma unroll
  for (int j = 0; j < 8; ++j) {
    float a = ok ? agg[(size_t)node * HDIM + q * 8 + j] : 0.f;
    sA[row * 288 + 128 + q * 8 + j] = f2bf(a);
  }
  if (q == 0) {
    float ext[32];
#pragma unroll
    for (int i = 0; i < 32; ++i) ext[i] = 0.f;
    if (ok) {
      ext[0]=dom[0]; ext[1]=dom[1]; ext[2]=dom[2]; ext[3]=t[0];
      ext[4]=xg[0]; ext[5]=xg[1]; ext[6]=xg[2]; ext[7]=xg[3];
      ext[8]=domn[0]; ext[9]=domn[1]; ext[10]=domn[2]; ext[11]=tn[0];
    }
#pragma unroll
    for (int i = 0; i < 32; ++i) sA[row * 288 + 256 + i] = f2bf(ext[i]);
  }
  wait_async_all();
  __syncthreads();
  const int wave = tid >> 5, lane = tid & 31;
  const int col = wave * 16 + (lane & 15);
  v8f acc = {};
  for (int kt = 0; kt < 9; ++kt) {
    v16bf a = a_frag_lds(sA, 288, kt, lane);
    v16bf b = b_frag_glb(W1 + (size_t)(kt * 8 + wave) * 512, lane);
    WMMA_BF16(acc, a, b);
  }
  float bias = b1[col];
#pragma unroll
  for (int i = 0; i < 8; ++i) {
    int rr = i + ((lane >> 4) << 3);
    sMid[rr * 128 + col] = f2bf(fmaxf(acc[i] + bias, 0.f));
  }
  __syncthreads();
  v8f acc2 = {};
  for (int kt = 0; kt < 4; ++kt) {
    v16bf a = a_frag_lds(sMid, 128, kt, lane);
    v16bf b = b_frag_glb(W2 + (size_t)(kt * 8 + wave) * 512, lane);
    WMMA_BF16(acc2, a, b);
  }
  float bias2 = b2[col];
  float ls = 0.f, lq = 0.f;
#pragma unroll
  for (int i = 0; i < 8; ++i) {
    int rr = i + ((lane >> 4) << 3);
    int nd = n0 + rr;
    float val = 0.f;
    if (nd < N_NODES) {
      val = h[(size_t)nd * HDIM + col] + acc2[i] + bias2;
      htmp[(size_t)nd * HDIM + col] = val;
    }
    ls += val; lq += val * val;
  }
  ls += __shfl_xor(ls, 16, 32);   // lanes L and L+16 hold the same column
  lq += __shfl_xor(lq, 16, 32);
  if (lane < 16) {
    atomicAdd(&sum[col], ls);
    atomicAdd(&sumsq[col], lq);
  }
}

__global__ void norm_kernel(const float* __restrict__ htmp,
                            const float* __restrict__ sum, const float* __restrict__ sumsq,
                            float* __restrict__ h, unsigned short* __restrict__ hbf,
                            float* __restrict__ agg_clear)
{
  size_t i = (size_t)blockIdx.x * blockDim.x + threadIdx.x;
  if (i >= (size_t)N_NODES * HDIM) return;
  int c = (int)(i & (HDIM - 1));
  const float inv = 1.0f / (float)N_NODES;
  float mean = sum[c] * inv;
  float var  = sumsq[c] * inv - mean * mean;
  float val  = (htmp[i] - mean) * rsqrtf(var + 1e-5f);
  h[i]   = val;
  hbf[i] = f2bf(val);
  if (agg_clear) agg_clear[i] = 0.f;   // zero agg for next layer
}

// ---------------------------------------------------------------- output head
__global__ __launch_bounds__(256) void out_kernel(
    const unsigned short* __restrict__ hbf,
    const float* __restrict__ pos, const float* __restrict__ v,
    const float* __restrict__ domn,
    const unsigned short* __restrict__ W1, const float* __restrict__ b1,
    const unsigned short* __restrict__ W2p, const float* __restrict__ b2,
    float* __restrict__ predx, float* __restrict__ predv)
{
  __shared__ __align__(16) unsigned short sH[16 * 128];
  __shared__ __align__(16) unsigned short sMid[16 * 128];
  const int tid = threadIdx.x;
  const int row = tid >> 4, q = tid & 15;
  const int n0 = blockIdx.x * 16;
  const int node = n0 + row;
  const int nodeC = (node < N_NODES) ? node : 0;
  async_ld_b128(&sH[row * 128 + q * 8], hbf + (size_t)nodeC * HDIM + q * 8);
  wait_async_all();
  __syncthreads();
  const int wave = tid >> 5, lane = tid & 31;
  const int col = wave * 16 + (lane & 15);
  v8f acc = {};
  for (int kt = 0; kt < 4; ++kt) {
    v16bf a = a_frag_lds(sH, 128, kt, lane);
    v16bf b = b_frag_glb(W1 + (size_t)(kt * 8 + wave) * 512, lane);
    WMMA_BF16(acc, a, b);
  }
  float bias = b1[col];
#pragma unroll
  for (int i = 0; i < 8; ++i) {
    int rr = i + ((lane >> 4) << 3);
    sMid[rr * 128 + col] = f2bf(fmaxf(acc[i] + bias, 0.f));
  }
  __syncthreads();
  if (wave == 0) {      // single 16-col tile (9 valid outputs)
    v8f acc2 = {};
    for (int kt = 0; kt < 4; ++kt) {
      v16bf a = a_frag_lds(sMid, 128, kt, lane);
      v16bf b = b_frag_glb(W2p + (size_t)kt * 512, lane);
      WMMA_BF16(acc2, a, b);
    }
    int oc = lane & 15;
#pragma unroll
    for (int i = 0; i < 8; ++i) {
      int rr = i + ((lane >> 4) << 3);
      int nd = n0 + rr;
      if (nd < N_NODES && oc < 9) {
        float o = acc2[i] + b2[oc];
        if (oc < 3) {
          float D = domn[oc];
          float p = 0.001f * o + pos[nd * 3 + oc];
          p = fmodf(p, D); if (p < 0.f) p += D;   // jnp.remainder semantics
          predx[nd * 3 + oc] = p;
        } else {
          int j = oc - 3;
          float sc = (oc < 6) ? 0.001f : 0.01f;   // SV / SW
          predv[nd * 6 + j] = sc * o + v[nd * 6 + j];
        }
      }
    }
  }
}

// ---------------------------------------------------------------- macro head
__global__ void colsum_kernel(const float* __restrict__ h, float* __restrict__ hsum) {
  int c = threadIdx.x & 127;
  int rbase = blockIdx.x * 2 + (threadIdx.x >> 7);
  float s = 0.f;
  for (int row = rbase; row < N_NODES; row += gridDim.x * 2)
    s += h[(size_t)row * HDIM + c];
  atomicAdd(&hsum[c], s);
}

__global__ void mac_kernel(const float* __restrict__ hsum,
                           const float* __restrict__ W1, const float* __restrict__ b1,
                           const float* __restrict__ W2, const float* __restrict__ b2,
                           float* __restrict__ outm) {
  __shared__ float hm[128];
  __shared__ float hid[128];
  int t = threadIdx.x;
  if (t < 128) hm[t] = hsum[t] / (float)N_NODES;
  __syncthreads();
  if (t < 128) {
    float s = b1[t];
    for (int c = 0; c < 128; ++c) s += hm[c] * W1[c * 128 + t];
    hid[t] = fmaxf(s, 0.f);
  }
  __syncthreads();
  if (t < 3) {
    float s = b2[t];
    for (int j = 0; j < 128; ++j) s += hid[j] * W2[j * 3 + t];
    outm[t] = s;
  }
}

// ---------------------------------------------------------------- launcher
extern "C" void kernel_launch(void* const* d_in, const int* in_sizes, int n_in,
                              void* d_out, int out_size, void* d_ws, size_t ws_size,
                              hipStream_t stream)
{
  (void)in_sizes; (void)n_in; (void)out_size; (void)ws_size;
  const float* vin   = (const float*)d_in[0];
  const float* posin = (const float*)d_in[1];
  const float* rin   = (const float*)d_in[2];
  const float* dom   = (const float*)d_in[3];
  const float* tt    = (const float*)d_in[4];
  const float* xg    = (const float*)d_in[5];
  const float* domn  = (const float*)d_in[6];
  const float* tn    = (const float*)d_in[7];
  const int*   ei    = (const int*)d_in[8];
  const float* embW1 = (const float*)d_in[10]; const float* embB1 = (const float*)d_in[11];
  const float* embW2 = (const float*)d_in[12]; const float* embB2 = (const float*)d_in[13];
  const float* msgW1 = (const float*)d_in[14]; const float* msgB1 = (const float*)d_in[15];
  const float* msgW2 = (const float*)d_in[16]; const float* msgB2 = (const float*)d_in[17];
  const float* updW1 = (const float*)d_in[18]; const float* updB1 = (const float*)d_in[19];
  const float* updW2 = (const float*)d_in[20]; const float* updB2 = (const float*)d_in[21];
  const float* outW1 = (const float*)d_in[22]; const float* outB1 = (const float*)d_in[23];
  const float* outW2 = (const float*)d_in[24]; const float* outB2 = (const float*)d_in[25];
  const float* macW1 = (const float*)d_in[26]; const float* macB1 = (const float*)d_in[27];
  const float* macW2 = (const float*)d_in[28]; const float* macB2 = (const float*)d_in[29];

  char* wsp = (char*)d_ws;
  auto bump = [&](size_t bytes) -> char* {
    char* p = wsp; wsp += (bytes + 255) & ~(size_t)255; return p;
  };
  float*          h     = (float*)bump((size_t)N_NODES * HDIM * 4);
  float*          htmp  = (float*)bump((size_t)N_NODES * HDIM * 4);
  unsigned short* hbf   = (unsigned short*)bump((size_t)N_NODES * HDIM * 2);
  unsigned short* wEmb1 = (unsigned short*)bump((size_t)32 * 128 * 2);
  unsigned short* wEmb2 = (unsigned short*)bump((size_t)128 * 128 * 2);
  unsigned short* wMsg1 = (unsigned short*)bump((size_t)NLAYER * 288 * 128 * 2);
  unsigned short* wMsg2 = (unsigned short*)bump((size_t)NLAYER * 128 * 128 * 2);
  unsigned short* wUpd1 = (unsigned short*)bump((size_t)NLAYER * 288 * 128 * 2);
  unsigned short* wUpd2 = (unsigned short*)bump((size_t)NLAYER * 128 * 128 * 2);
  unsigned short* wOut1 = (unsigned short*)bump((size_t)128 * 128 * 2);
  unsigned short* wOut2 = (unsigned short*)bump((size_t)128 * 16 * 2);
  char* zbase = wsp;                               // zero-init region starts here
  float* agg  = (float*)bump((size_t)N_NODES * HDIM * 4);
  float* deg  = (float*)bump((size_t)N_NODES * 4);
  float* sums = (float*)bump((size_t)NLAYER * 256 * 4);   // per layer: [sum(128) | sumsq(128)]
  float* hsum = (float*)bump(128 * 4);
  size_t zcount = (size_t)(wsp - zbase) / 4;

  zero_f32<<<(int)((zcount + 255) / 256), 256, 0, stream>>>((float*)zbase, zcount);

  auto convw = [&](const float* src, unsigned short* dst, int K, int Nn, int Kp, int Np) {
    int total = Kp * Np;
    conv_w<<<(total + 255) / 256, 256, 0, stream>>>(src, dst, K, Nn, Kp, Np);
  };
  convw(embW1, wEmb1, 19, 128, 32, 128);
  convw(embW2, wEmb2, 128, 128, 128, 128);
  for (int l = 0; l < NLAYER; ++l) {
    convw(msgW1 + (size_t)l * 279 * 128, wMsg1 + (size_t)l * 288 * 128, 279, 128, 288, 128);
    convw(msgW2 + (size_t)l * 128 * 128, wMsg2 + (size_t)l * 128 * 128, 128, 128, 128, 128);
    convw(updW1 + (size_t)l * 268 * 128, wUpd1 + (size_t)l * 288 * 128, 268, 128, 288, 128);
    convw(updW2 + (size_t)l * 128 * 128, wUpd2 + (size_t)l * 128 * 128, 128, 128, 128, 128);
  }
  convw(outW1, wOut1, 128, 128, 128, 128);
  convw(outW2, wOut2, 128, 9, 128, 16);

  embed_kernel<<<(N_NODES + 15) / 16, 256, 0, stream>>>(
      rin, vin, dom, tt, xg, domn, tn, wEmb1, embB1, wEmb2, embB2, h, hbf);
  deg_kernel<<<(N_EDGES + 255) / 256, 256, 0, stream>>>(ei, deg);

  for (int l = 0; l < NLAYER; ++l) {
    edge_kernel<<<(N_EDGES + 15) / 16, 256, 0, stream>>>(
        ei, hbf, posin, vin, rin, dom, tt, xg, domn, tn,
        wMsg1 + (size_t)l * 288 * 128, msgB1 + (size_t)l * 128,
        wMsg2 + (size_t)l * 128 * 128, msgB2 + (size_t)l * 128, deg, agg);
    node_kernel<<<(N_NODES + 15) / 16, 256, 0, stream>>>(
        hbf, h, agg, dom, tt, xg, domn, tn,
        wUpd1 + (size_t)l * 288 * 128, updB1 + (size_t)l * 128,
        wUpd2 + (size_t)l * 128 * 128, updB2 + (size_t)l * 128,
        htmp, sums + (size_t)l * 256, sums + (size_t)l * 256 + 128);
    norm_kernel<<<(int)(((size_t)N_NODES * HDIM + 255) / 256), 256, 0, stream>>>(
        htmp, sums + (size_t)l * 256, sums + (size_t)l * 256 + 128,
        h, hbf, (l < NLAYER - 1) ? agg : nullptr);
  }

  float* predx = (float*)d_out;
  float* predv = (float*)d_out + (size_t)N_NODES * 3;
  float* predm = (float*)d_out + (size_t)N_NODES * 9;
  out_kernel<<<(N_NODES + 15) / 16, 256, 0, stream>>>(
      hbf, posin, vin, domn, wOut1, outB1, wOut2, outB2, predx, predv);
  colsum_kernel<<<512, 256, 0, stream>>>(h, hsum);
  mac_kernel<<<1, 128, 0, stream>>>(hsum, macW1, macB1, macW2, macB2, predm);
}
